// Attention_49581102465518
// MI455X (gfx1250) — matrix operand order repaired and checked by reading, not compiled, not run on hardware
//
#include <hip/hip_runtime.h>
#include <hip/hip_bf16.h>

#define H_   16
#define KVH_ 4
#define D_   2048
#define HD_  128
#define S_   4096
#define B_   2
#define NQK  3072          // wq(2048) | wk(512) | wv(512) concatenated columns
#define BS_  (B_ * S_)     // 8192 rows

typedef __attribute__((ext_vector_type(16))) __bf16 v16bf;
typedef __attribute__((ext_vector_type(8)))  __bf16 v8bf;
typedef __attribute__((ext_vector_type(8)))  float  v8f;
typedef __attribute__((ext_vector_type(4)))  float  v4f;

static __device__ __forceinline__ v16bf frag_cat(v8bf lo, v8bf hi) {
    v16bf r;
#pragma unroll
    for (int i = 0; i < 8; ++i) { r[i] = lo[i]; r[i + 8] = hi[i]; }
    return r;
}

// A fragment (16x32 bf16), tile row-major [row][k], ld in elements.
// Lane layout: row = lane&15; comps 0..7 -> K = koff + 8*half + j,
// comps 8..15 -> K = koff + 16 + 8*half + j.
static __device__ __forceinline__ v16bf load_a_frag(const __bf16* t, int ld, int row,
                                                    int koff, int half) {
    const __bf16* p = t + row * ld + koff + 8 * half;
    return frag_cat(*(const v8bf*)p, *(const v8bf*)(p + 16));
}

// B fragment (32x16 bf16), tile stored TRANSPOSED [n][k], ld in elements.
// Lane layout: col = lane&15; comps 0..7 -> K = koff + 8*half + j,
// comps 8..15 -> K = koff + 16 + 8*half + j.
static __device__ __forceinline__ v16bf load_b_frag(const __bf16* t, int ld, int col,
                                                    int koff, int half) {
    const __bf16* p = t + col * ld + koff + 8 * half;
    return frag_cat(*(const v8bf*)p, *(const v8bf*)(p + 16));
}

static __device__ __forceinline__ v8f wmma_bf16(v16bf a, v16bf b, v8f c) {
    return __builtin_amdgcn_wmma_f32_16x16x32_bf16(false, a, false, b, (short)0, c,
                                                   false, false);
}

// pack 4 fp32 -> 4 bf16 and store as one 8-byte LDS write
static __device__ __forceinline__ void store4_bf16(__bf16* p, v4f v) {
    union { __bf16 b[4]; unsigned long long u; } pk;
    pk.b[0] = (__bf16)v[0]; pk.b[1] = (__bf16)v[1];
    pk.b[2] = (__bf16)v[2]; pk.b[3] = (__bf16)v[3];
    *(unsigned long long*)p = pk.u;
}

// ---------------------------------------------------------------------------
// Kernel 1: fused QKV projection.  raw[8192][3072] = x[8192][2048] @ [wq|wk|wv]
// Block: 256 threads (8 waves), tile 128M x 128N, K-step 32.
// Wave grid 4x2: each wave 32M x 64N = 2x4 wmma tiles, 8 wmma / k-iter.
// ---------------------------------------------------------------------------
__global__ __launch_bounds__(256) void qkv_gemm_kernel(
    const float* __restrict__ x, const float* __restrict__ wq,
    const float* __restrict__ wk, const float* __restrict__ wv,
    float* __restrict__ raw) {
    __shared__ __align__(16) __bf16 As[128][32];   // [m][k]
    __shared__ __align__(16) __bf16 Bs[128][32];   // [n][k] (transposed)
    const int tid = threadIdx.x;
    const int m0  = blockIdx.x * 128;
    const int n0  = blockIdx.y * 128;      // global col 0..3071, tile-aligned to weight
    const float* w; int wld, c0;
    if (n0 < D_)            { w = wq; wld = D_;  c0 = n0; }
    else if (n0 < D_ + 512) { w = wk; wld = 512; c0 = n0 - D_; }
    else                    { w = wv; wld = 512; c0 = n0 - D_ - 512; }

    const int wave = tid >> 5, lane = tid & 31;
    const int wm = wave >> 1, wn = wave & 1;       // 4x2 wave grid
    const int lrow = lane & 15, half = lane >> 4;

    // staging assignments (vectorized, coalesced)
    const int am = tid >> 3, ak = (tid & 7) << 2;        // A: 8 float4 per row
    const int bk = tid >> 5, bn = (tid & 31) << 2;       // B: 32 float4 per k-row
    const float* xa = x + (size_t)(m0 + am) * D_ + ak;   // += 32 per iter
    const float* wa = w + (size_t)bk * wld + c0 + bn;    // += 32*wld per iter

    v8f acc[2][4];
#pragma unroll
    for (int i = 0; i < 2; ++i)
#pragma unroll
        for (int j = 0; j < 4; ++j) acc[i][j] = (v8f){0, 0, 0, 0, 0, 0, 0, 0};

    for (int k0 = 0; k0 < D_; k0 += 32) {
#pragma unroll
        for (int i = 0; i < 4; ++i) {              // A: 128x32 fp32 -> bf16
            v4f v = *(const v4f*)(xa + (size_t)i * 32 * D_);
            store4_bf16(&As[am + i * 32][ak], v);
        }
#pragma unroll
        for (int i = 0; i < 4; ++i) {              // B: 32x128 fp32 -> bf16^T
            v4f v = *(const v4f*)(wa + (size_t)i * 8 * wld);
            const int k = bk + i * 8;
#pragma unroll
            for (int e = 0; e < 4; ++e) Bs[bn + e][k] = (__bf16)v[e];
        }
        if (k0 + 32 < D_) {                        // global_prefetch_b8 next tiles
            __builtin_prefetch(xa + 32, 0, 1);
            __builtin_prefetch(wa + (size_t)32 * wld, 0, 1);
        }
        xa += 32;
        wa += (size_t)32 * wld;
        __syncthreads();
        v16bf af[2], bf2[4];
#pragma unroll
        for (int tm = 0; tm < 2; ++tm)
            af[tm] = load_a_frag(&As[0][0], 32, wm * 32 + tm * 16 + lrow, 0, half);
#pragma unroll
        for (int tn = 0; tn < 4; ++tn)
            bf2[tn] = load_b_frag(&Bs[0][0], 32, wn * 64 + tn * 16 + lrow, 0, half);
#pragma unroll
        for (int tm = 0; tm < 2; ++tm)
#pragma unroll
            for (int tn = 0; tn < 4; ++tn)
                acc[tm][tn] = wmma_bf16(af[tm], bf2[tn], acc[tm][tn]);
        __syncthreads();
    }
    // C layout: row = 8*half + j, col = lane&15 within each 16x16 tile
#pragma unroll
    for (int tm = 0; tm < 2; ++tm)
#pragma unroll
        for (int tn = 0; tn < 4; ++tn)
#pragma unroll
            for (int j = 0; j < 8; ++j) {
                int m = m0 + wm * 32 + tm * 16 + 8 * half + j;
                int n = n0 + wn * 64 + tn * 16 + lrow;
                raw[(size_t)m * NQK + n] = acc[tm][tn][j];
            }
}

// ---------------------------------------------------------------------------
// Kernel 2: RoPE + head transpose + bf16 pack.
// dst[b][h][s][d] = rope(raw[b*S+s][col0 + h*128 + d])
// ---------------------------------------------------------------------------
__global__ __launch_bounds__(256) void rope_pack_kernel(
    const float* __restrict__ raw, const float* __restrict__ cosb,
    const float* __restrict__ sinb, __bf16* __restrict__ dst, int heads, int col0,
    int do_rope, long total) {
    long idx = (long)blockIdx.x * 256 + threadIdx.x;
    if (idx >= total) return;
    int d = (int)(idx & (HD_ - 1));
    long t = idx >> 7;
    int s = (int)(t & (S_ - 1)); t >>= 12;
    int h = (int)(t % heads);
    int b = (int)(t / heads);
    size_t row = (size_t)(b * S_ + s);
    size_t off = row * NQK + col0 + h * HD_;
    float v = raw[off + d];
    float o = v;
    if (do_rope) {
        float c  = cosb[row * HD_ + d];
        float sn = sinb[row * HD_ + d];
        float other = (d < 64) ? -raw[off + d + 64] : raw[off + d - 64];
        o = v * c + other * sn;
    }
    dst[(((size_t)(b * heads + h)) * S_ + s) * HD_ + d] = (__bf16)o;
}

// ---------------------------------------------------------------------------
// Kernel 3: causal flash attention with GQA (4 q-heads per kv-head).
// Block = 128 threads (4 waves); each block: 64 q rows of one (b,h).
// ---------------------------------------------------------------------------
__global__ __launch_bounds__(128) void flash_attn_kernel(
    const __bf16* __restrict__ qb, const __bf16* __restrict__ kb,
    const __bf16* __restrict__ vb, __bf16* __restrict__ attn) {
    __shared__ __align__(16) __bf16 Qs[64][128];   // [q][d]     (A tiles)
    __shared__ __align__(16) __bf16 Ks[64][128];   // [kv][d]    (B tiles for QK^T)
    __shared__ __align__(16) __bf16 Vs[128][64];   // [d][kv]    (B tiles for PV)
    __shared__ __align__(16) __bf16 Ps[4][16][64]; // per-wave P (A tiles for PV)
    const int tid = threadIdx.x;
    const int wave = tid >> 5, lane = tid & 31;
    const int lrow = lane & 15, half = lane >> 4;
    int blk = blockIdx.x;
    const int qt = blk & 63; blk >>= 6;
    const int h = blk & (H_ - 1);
    const int b = blk >> 4;
    const int hk = h / (H_ / KVH_);
    const size_t qoff = (((size_t)(b * H_ + h)) * S_ + qt * 64) * HD_;
    const size_t kof  = ((size_t)(b * KVH_ + hk)) * S_ * HD_;

    {   // load Q tile once (64x128 bf16), 16B vector copies
        const int r = tid & 63, hh = tid >> 6;
        const uint4* src = (const uint4*)(qb + qoff + (size_t)r * HD_ + hh * 64);
        uint4* dp = (uint4*)(&Qs[r][hh * 64]);
#pragma unroll
        for (int i = 0; i < 8; ++i) dp[i] = src[i];
    }

    float m_i[8], l_i[8];
    v8f oacc[8];
#pragma unroll
    for (int j = 0; j < 8; ++j) { m_i[j] = -__builtin_inff(); l_i[j] = 0.f; }
#pragma unroll
    for (int c = 0; c < 8; ++c) oacc[c] = (v8f){0, 0, 0, 0, 0, 0, 0, 0};

    const float scale = 0.08838834764831845f;  // 1/sqrt(128)

    for (int jv = 0; jv <= qt; ++jv) {
        __syncthreads();
        {   // stage K tile row-major, V tile transposed
            const int r = tid & 63, hh = tid >> 6;
            const size_t kvrow = kof + (size_t)(jv * 64 + r) * HD_ + hh * 64;
            const uint4* ks = (const uint4*)(kb + kvrow);
            uint4* kd = (uint4*)(&Ks[r][hh * 64]);
#pragma unroll
            for (int i = 0; i < 8; ++i) kd[i] = ks[i];
            const v8bf* vsrc = (const v8bf*)(vb + kvrow);
#pragma unroll
            for (int i = 0; i < 8; ++i) {
                v8bf vv = vsrc[i];
#pragma unroll
                for (int e = 0; e < 8; ++e) Vs[hh * 64 + i * 8 + e][r] = vv[e];
            }
        }
        __syncthreads();

        // S = Q K^T  (16 q rows x 64 kv per wave)
        v8f sacc[4];
#pragma unroll
        for (int c = 0; c < 4; ++c) sacc[c] = (v8f){0, 0, 0, 0, 0, 0, 0, 0};
        v16bf aq[4];
#pragma unroll
        for (int kc = 0; kc < 4; ++kc)
            aq[kc] = load_a_frag(&Qs[0][0], HD_, wave * 16 + lrow, kc * 32, half);
#pragma unroll
        for (int c = 0; c < 4; ++c)
#pragma unroll
            for (int kc = 0; kc < 4; ++kc) {
                v16bf bk = load_b_frag(&Ks[0][0], HD_, c * 16 + lrow, kc * 32, half);
                sacc[c] = wmma_bf16(aq[kc], bk, sacc[c]);
            }

        // online softmax (rows live in comps j, halves of the wave)
        float sv[4][8];
        const int rowg0 = qt * 64 + wave * 16 + 8 * half;
        const bool diag = (jv == qt);
#pragma unroll
        for (int c = 0; c < 4; ++c)
#pragma unroll
            for (int j = 0; j < 8; ++j) {
                float s = sacc[c][j] * scale;
                if (diag) {
                    int col = jv * 64 + c * 16 + lrow;
                    if (col > rowg0 + j) s = -__builtin_inff();
                }
                sv[c][j] = s;
            }
        float mx[8];
#pragma unroll
        for (int j = 0; j < 8; ++j) {
            float m = fmaxf(fmaxf(sv[0][j], sv[1][j]), fmaxf(sv[2][j], sv[3][j]));
#pragma unroll
            for (int off = 8; off >= 1; off >>= 1)
                m = fmaxf(m, __shfl_xor(m, off, 32));   // stays within 16-lane half
            mx[j] = m;
        }
        float alpha[8];
#pragma unroll
        for (int j = 0; j < 8; ++j) {
            float mn = fmaxf(m_i[j], mx[j]);
            alpha[j] = __expf(m_i[j] - mn);
            m_i[j] = mn;
        }
        float rs[8] = {0, 0, 0, 0, 0, 0, 0, 0};
#pragma unroll
        for (int c = 0; c < 4; ++c)
#pragma unroll
            for (int j = 0; j < 8; ++j) {
                float p = __expf(sv[c][j] - m_i[j]);
                sv[c][j] = p;
                rs[j] += p;
            }
#pragma unroll
        for (int j = 0; j < 8; ++j) {
            float r = rs[j];
#pragma unroll
            for (int off = 8; off >= 1; off >>= 1) r += __shfl_xor(r, off, 32);
            l_i[j] = l_i[j] * alpha[j] + r;
        }
#pragma unroll
        for (int c = 0; c < 8; ++c)
#pragma unroll
            for (int j = 0; j < 8; ++j) oacc[c][j] *= alpha[j];

        // P -> per-wave LDS (C layout -> row-major), then reuse as A fragment
#pragma unroll
        for (int c = 0; c < 4; ++c)
#pragma unroll
            for (int j = 0; j < 8; ++j)
                Ps[wave][8 * half + j][c * 16 + lrow] = (__bf16)sv[c][j];
        asm volatile("s_wait_dscnt 0x0" ::: "memory");  // wave-local LDS RAW fence

        // O += P @ V   (16x64 @ 64x128)
#pragma unroll
        for (int kc = 0; kc < 2; ++kc) {
            v16bf ap = load_a_frag(&Ps[wave][0][0], 64, lrow, kc * 32, half);
#pragma unroll
            for (int cn = 0; cn < 8; ++cn) {
                v16bf bv = load_b_frag(&Vs[0][0], 64, cn * 16 + lrow, kc * 32, half);
                oacc[cn] = wmma_bf16(ap, bv, oacc[cn]);
            }
        }
    }

    // epilogue: normalize and write attn[b][s][h*128 + d] (bf16)
#pragma unroll
    for (int cn = 0; cn < 8; ++cn)
#pragma unroll
        for (int j = 0; j < 8; ++j) {
            int row = qt * 64 + wave * 16 + 8 * half + j;
            int col = cn * 16 + lrow;
            float o = oacc[cn][j] / l_i[j];
            attn[((size_t)(b * S_) + row) * D_ + h * HD_ + col] = (__bf16)o;
        }
}

// ---------------------------------------------------------------------------
// Kernel 4: output projection.  out[8192][2048] = attn(bf16) @ wo
// Same tiling as kernel 1 (128x128, 8 waves, 8 wmma/iter).
// ---------------------------------------------------------------------------
__global__ __launch_bounds__(256) void out_proj_kernel(
    const __bf16* __restrict__ a, const float* __restrict__ wo,
    float* __restrict__ out) {
    __shared__ __align__(16) __bf16 As[128][32];
    __shared__ __align__(16) __bf16 Bs[128][32];
    const int tid = threadIdx.x;
    const int m0 = blockIdx.x * 128;
    const int n0 = blockIdx.y * 128;
    const int wave = tid >> 5, lane = tid & 31;
    const int wm = wave >> 1, wn = wave & 1;
    const int lrow = lane & 15, half = lane >> 4;

    // A (bf16): 128x32 = 512 uint4 vectors -> 2 per thread
    const int am = tid >> 2, ak = (tid & 3) << 3;
    const __bf16* aa = a + (size_t)(m0 + am) * D_ + ak;      // += 32 per iter
    const int bk = tid >> 5, bn = (tid & 31) << 2;
    const float* wa = wo + (size_t)bk * D_ + n0 + bn;        // += 32*D_ per iter

    v8f acc[2][4];
#pragma unroll
    for (int i = 0; i < 2; ++i)
#pragma unroll
        for (int j = 0; j < 4; ++j) acc[i][j] = (v8f){0, 0, 0, 0, 0, 0, 0, 0};

    for (int k0 = 0; k0 < D_; k0 += 32) {
#pragma unroll
        for (int i = 0; i < 2; ++i) {              // A already bf16: 16B copies
            *(uint4*)(&As[am + i * 64][ak]) =
                *(const uint4*)(aa + (size_t)i * 64 * D_);
        }
#pragma unroll
        for (int i = 0; i < 4; ++i) {              // B: 32x128 fp32 -> bf16^T
            v4f v = *(const v4f*)(wa + (size_t)i * 8 * D_);
            const int k = bk + i * 8;
#pragma unroll
            for (int e = 0; e < 4; ++e) Bs[bn + e][k] = (__bf16)v[e];
        }
        if (k0 + 32 < D_) {
            __builtin_prefetch(aa + 32, 0, 1);
            __builtin_prefetch(wa + (size_t)32 * D_, 0, 1);
        }
        aa += 32;
        wa += (size_t)32 * D_;
        __syncthreads();
        v16bf af[2], bf2[4];
#pragma unroll
        for (int tm = 0; tm < 2; ++tm)
            af[tm] = load_a_frag(&As[0][0], 32, wm * 32 + tm * 16 + lrow, 0, half);
#pragma unroll
        for (int tn = 0; tn < 4; ++tn)
            bf2[tn] = load_b_frag(&Bs[0][0], 32, wn * 64 + tn * 16 + lrow, 0, half);
#pragma unroll
        for (int tm = 0; tm < 2; ++tm)
#pragma unroll
            for (int tn = 0; tn < 4; ++tn)
                acc[tm][tn] = wmma_bf16(af[tm], bf2[tn], acc[tm][tn]);
        __syncthreads();
    }
#pragma unroll
    for (int tm = 0; tm < 2; ++tm)
#pragma unroll
        for (int tn = 0; tn < 4; ++tn)
#pragma unroll
            for (int j = 0; j < 8; ++j) {
                int m = m0 + wm * 32 + tm * 16 + 8 * half + j;
                int n = n0 + wn * 64 + tn * 16 + lrow;
                out[(size_t)m * D_ + n] = acc[tm][tn][j];
            }
}

// ---------------------------------------------------------------------------
extern "C" void kernel_launch(void* const* d_in, const int* in_sizes, int n_in,
                              void* d_out, int out_size, void* d_ws, size_t ws_size,
                              hipStream_t stream) {
    (void)in_sizes; (void)n_in; (void)out_size; (void)ws_size;
    const float* x    = (const float*)d_in[0];
    const float* cosb = (const float*)d_in[1];
    const float* sinb = (const float*)d_in[2];
    const float* wq   = (const float*)d_in[3];
    const float* wk   = (const float*)d_in[4];
    const float* wv   = (const float*)d_in[5];
    const float* wo   = (const float*)d_in[6];
    float* out = (float*)d_out;

    // workspace layout (151 MB):
    //   [0, 100663296)            raw QKV fp32 [8192][3072]   (later aliased by attn bf16)
    //   [+0]                      attn bf16   [8192][2048]    (raw is dead by then)
    //   [100663296, +33554432)    qb bf16 [B][H][S][HD]
    //   [+, +8388608)             kb bf16 [B][KVH][S][HD]
    //   [+, +8388608)             vb bf16 [B][KVH][S][HD]
    float*  raw  = (float*)d_ws;
    __bf16* qb   = (__bf16*)((char*)d_ws + (size_t)BS_ * NQK * 4);
    __bf16* kb   = qb + (size_t)B_ * H_ * S_ * HD_;
    __bf16* vb   = kb + (size_t)B_ * KVH_ * S_ * HD_;
    __bf16* attn = (__bf16*)d_ws;

    qkv_gemm_kernel<<<dim3(BS_ / 128, NQK / 128), 256, 0, stream>>>(x, wq, wk, wv, raw);

    const long qtot = (long)B_ * H_ * S_ * HD_;
    const long ktot = (long)B_ * KVH_ * S_ * HD_;
    rope_pack_kernel<<<(int)((qtot + 255) / 256), 256, 0, stream>>>(
        raw, cosb, sinb, qb, H_, 0, 1, qtot);
    rope_pack_kernel<<<(int)((ktot + 255) / 256), 256, 0, stream>>>(
        raw, cosb, sinb, kb, KVH_, D_, 1, ktot);
    rope_pack_kernel<<<(int)((ktot + 255) / 256), 256, 0, stream>>>(
        raw, cosb, sinb, vb, KVH_, D_ + 512, 0, ktot);

    flash_attn_kernel<<<B_ * H_ * (S_ / 64), 128, 0, stream>>>(qb, kb, vb, attn);

    out_proj_kernel<<<dim3(BS_ / 128, D_ / 128), 256, 0, stream>>>(attn, wo, out);
}
